// MyRNN_38817914421499
// MI455X (gfx1250) — matrix operand order repaired and checked by reading, not compiled, not run on hardware
//
#include <hip/hip_runtime.h>
#include <hip/hip_bf16.h>

// ---------------------------------------------------------------------------
// 3-layer SimpleRNN on MI455X (gfx1250), wave32 + WMMA bf16 + async LDS DMA.
//
// 8 batch-blocks (32 rows) x 8 column-slices (64 cols) = 64 persistent WGs.
// Each WG keeps its 208KB weight column-slice resident in LDS (bf16, stored in
// WMMA B-fragment order) for all 512 timesteps. Hidden states are exchanged
// between the 8 slice-WGs of a batch block via double-buffered global bf16
// buffers, staged into LDS with GLOBAL_LOAD_ASYNC_TO_LDS_B128 (ASYNCcnt), and
// synchronized with an 8-member monotonic atomic barrier (3/timestep).
// Matmuls: v_wmma_f32_16x16x32_bf16, f32 accumulate, tanh epilogue in f32.
// ---------------------------------------------------------------------------

#define BATCH 256
#define SEQ   512
#define EMB   128
#define UNIT  512

#define BBLOCKS   8     // batch blocks of 32 rows
#define NSLICES   8     // column slices of 64 columns
#define ROWS_WG   32
#define COLS_WG   64
#define KT_E      (EMB / 32)    // 4
#define KT_U      (UNIT / 32)   // 16
#define WG_THREADS 256          // 8 waves

typedef __attribute__((ext_vector_type(16))) __bf16 v16bf;
typedef __attribute__((ext_vector_type(8)))  float  v8f;

// LDS layout (ushort units)
#define L_WX0   0                       // 4*KT_E*512 = 8192
#define L_WH0   (L_WX0 + 4*KT_E*512)    // +32768
#define L_WX1   (L_WH0 + 4*KT_U*512)
#define L_WH1   (L_WX1 + 4*KT_U*512)
#define L_XT    (L_WH1 + 4*KT_U*512)    // 32*128 = 4096
#define L_A     (L_XT + ROWS_WG*EMB)    // 32*512 = 16384
#define L_B     (L_A + ROWS_WG*UNIT)    // 16384
#define L_TOTAL (L_B + ROWS_WG*UNIT)    // 143360 ushorts
#define SMEM_BYTES (L_TOTAL * 2)        // 286720 B < 320KB/WGP

__device__ __forceinline__ unsigned short f2bf(float f) {
    unsigned u = __float_as_uint(f);
    return (unsigned short)((u + 0x7FFFu + ((u >> 16) & 1u)) >> 16);
}
__device__ __forceinline__ float bf2f(unsigned short h) {
    return __uint_as_float(((unsigned)h) << 16);
}

__device__ __forceinline__ v8f wmma_bf16(v16bf a, v16bf b, v8f c) {
    return __builtin_amdgcn_wmma_f32_16x16x32_bf16(
        false, a, false, b, (short)0, c, false, false);
}

// A fragment (16x32 bf16) from row-major bf16 LDS, ISA 16-bit A layout:
// lane m=lane&15, hh=lane>>4; VGPR r holds K pair Kloc = 2r + (r>=4?8:0) + 8*hh.
// Two 16B-aligned contiguous groups -> force 2x ds_load_b128 via uint4.
__device__ __forceinline__ v16bf ldA(const unsigned short* buf, int stride,
                                     int row, int hh, int k0) {
    union { v16bf v; uint4 q[2]; } A;
    const uint4* p =
        (const uint4*)((const unsigned*)(buf + row * stride + k0) + 4 * hh);
    A.q[0] = p[0];
    A.q[1] = p[2];
    return A.v;
}

// B fragment: pre-swizzled in memory, lane-contiguous 32B.
__device__ __forceinline__ v16bf ldB(const unsigned short* wLane, int kt) {
    return *(const v16bf*)(wLane + kt * 512);
}

// Async global->LDS copy: per-lane 16B chunks, GLOBAL_LOAD_ASYNC_TO_LDS_B128.
// ldsBytes is the absolute LDS byte offset of the destination.
__device__ __forceinline__ void copy_async16(unsigned ldsBytes,
                                             const unsigned short* src,
                                             int nels, int tid) {
    int n = nels >> 3;   // 16-byte chunks
    for (int i = tid; i < n; i += WG_THREADS) {
        unsigned dst = ldsBytes + (unsigned)i * 16u;
        unsigned long long ga = (unsigned long long)(size_t)(src + i * 8);
        asm volatile("global_load_async_to_lds_b128 %0, %1, off"
                     :: "v"(dst), "v"(ga) : "memory");
    }
}
__device__ __forceinline__ void wait_async0() {
    asm volatile("s_wait_asynccnt 0x0" ::: "memory");
}

// Monotonic 8-member barrier (one counter per batch block).
__device__ __forceinline__ void group_barrier(unsigned int* c, unsigned int phase) {
    __threadfence();
    __syncthreads();
    if (threadIdx.x == 0) {
        __hip_atomic_fetch_add(c, 1u, __ATOMIC_ACQ_REL, __HIP_MEMORY_SCOPE_AGENT);
        while (__hip_atomic_load(c, __ATOMIC_ACQUIRE, __HIP_MEMORY_SCOPE_AGENT)
               < (unsigned)NSLICES * phase) {
            __builtin_amdgcn_s_sleep(2);
        }
    }
    __syncthreads();
    __threadfence();
}

// One RNN cell: D = tanh(A1@W1 + A2@W2 + bias), A2 always [32,512].
template <int KT1>
__device__ __forceinline__ void stage_compute(
    const unsigned short* a1, int a1Stride, const unsigned short* w1Lane,
    const unsigned short* a2, const unsigned short* w2Lane,
    int aRow, int hh, unsigned short* dst, int rowBase, int mt, int j,
    float bias) {
    v8f ce = {0, 0, 0, 0, 0, 0, 0, 0};
    v8f co = {0, 0, 0, 0, 0, 0, 0, 0};
#pragma unroll
    for (int kt = 0; kt < KT1; kt += 2) {
        ce = wmma_bf16(ldA(a1, a1Stride, aRow, hh, kt * 32),      ldB(w1Lane, kt),     ce);
        co = wmma_bf16(ldA(a1, a1Stride, aRow, hh, kt * 32 + 32), ldB(w1Lane, kt + 1), co);
    }
#pragma unroll
    for (int kt = 0; kt < KT_U; kt += 2) {
        ce = wmma_bf16(ldA(a2, UNIT, aRow, hh, kt * 32),      ldB(w2Lane, kt),     ce);
        co = wmma_bf16(ldA(a2, UNIT, aRow, hh, kt * 32 + 32), ldB(w2Lane, kt + 1), co);
    }
    v8f cs = ce + co;
#pragma unroll
    for (int r = 0; r < 8; ++r) {
        float x = tanhf(cs[r] + bias);
        dst[(rowBase + mt * 16 + 8 * hh + r) * UNIT + j] = f2bf(x);
    }
}

// Prologue: f32 weights -> bf16 fragments ordered [nt][kt][lane*16+e],
// B layout: N = nt*16 + lane%16, K = kt*32 + 8*(lane>=16) + (e&7) + 16*(e>>3).
__global__ void convert_weights(const float* __restrict__ W,
                                unsigned short* __restrict__ out, int K) {
    int f = blockIdx.x * 256 + threadIdx.x;
    if (f >= K * 512) return;
    int e    = f & 15;
    int lane = (f >> 4) & 31;
    int tile = f >> 9;
    int KT   = K >> 5;
    int kt   = tile % KT;
    int nt   = tile / KT;
    int kk   = kt * 32 + 8 * (lane >> 4) + (e & 7) + 16 * (e >> 3);
    int n    = nt * 16 + (lane & 15);
    out[f] = f2bf(W[kk * UNIT + n]);
}

__global__ void zero_ws(unsigned int* __restrict__ p, int n) {
    int i = blockIdx.x * 256 + threadIdx.x;
    if (i < n) p[i] = 0u;
}

__global__ __launch_bounds__(WG_THREADS, 1)
void rnn_main(const int* __restrict__ tokens, const float* __restrict__ emb,
              const float* __restrict__ b0, const float* __restrict__ b1,
              const float* __restrict__ fcw, const float* __restrict__ fcb,
              const unsigned short* __restrict__ wfx0,
              const unsigned short* __restrict__ wfh0,
              const unsigned short* __restrict__ wfx1,
              const unsigned short* __restrict__ wfh1,
              unsigned short* __restrict__ hbuf,   // [3][2][256][512] bf16
              unsigned int* __restrict__ cnt,      // [BBLOCKS]
              float* __restrict__ out) {
    extern __shared__ unsigned short smem[];
    unsigned short* s_wx0 = smem + L_WX0;
    unsigned short* s_wh0 = smem + L_WH0;
    unsigned short* s_wx1 = smem + L_WX1;
    unsigned short* s_wh1 = smem + L_WH1;
    unsigned short* s_xt  = smem + L_XT;
    unsigned short* s_A   = smem + L_A;
    unsigned short* s_B   = smem + L_B;

    // Absolute LDS byte offsets (dynamic LDS starts after static LDS == 0 here).
    const unsigned ldsBase = __builtin_amdgcn_groupstaticsize();
    const unsigned bWX0 = ldsBase + L_WX0 * 2;
    const unsigned bWH0 = ldsBase + L_WH0 * 2;
    const unsigned bWX1 = ldsBase + L_WX1 * 2;
    const unsigned bWH1 = ldsBase + L_WH1 * 2;
    const unsigned bA   = ldsBase + L_A * 2;
    const unsigned bB   = ldsBase + L_B * 2;

    const int tid  = threadIdx.x;
    const int bb   = blockIdx.x >> 3;
    const int sl   = blockIdx.x & 7;
    const int rowBase = bb * ROWS_WG;
    const int colBase = sl * COLS_WG;

    // Resident weight slices (async DMA once, wait folds into t=0).
    copy_async16(bWX0, wfx0 + (size_t)sl * 4 * KT_E * 512, 4 * KT_E * 512, tid);
    copy_async16(bWH0, wfh0 + (size_t)sl * 4 * KT_U * 512, 4 * KT_U * 512, tid);
    copy_async16(bWX1, wfx1 + (size_t)sl * 4 * KT_U * 512, 4 * KT_U * 512, tid);
    copy_async16(bWH1, wfh1 + (size_t)sl * 4 * KT_U * 512, 4 * KT_U * 512, tid);

    const int lane = tid & 31;
    const int wave = tid >> 5;
    const int mt   = wave >> 2;       // M tile (0..1)
    const int ntL  = wave & 3;        // local N tile (0..3)
    const int hh   = lane >> 4;
    const int m    = lane & 15;
    const int aRow = mt * 16 + m;
    const int j    = colBase + ntL * 16 + m;   // output column (C/D: N = lane%16)

    const float bias0 = b0[j];
    const float bias1 = b1[j];

    const unsigned short* wl_x0 = s_wx0 + (ntL * KT_E) * 512 + lane * 16;
    const unsigned short* wl_h0 = s_wh0 + (ntL * KT_U) * 512 + lane * 16;
    const unsigned short* wl_x1 = s_wx1 + (ntL * KT_U) * 512 + lane * 16;
    const unsigned short* wl_h1 = s_wh1 + (ntL * KT_U) * 512 + lane * 16;

    unsigned short* h0b = hbuf;                         // [2][256][512]
    unsigned short* h1b = hbuf + 2 * BATCH * UNIT;
    unsigned short* h2b = hbuf + 4 * BATCH * UNIT;
    unsigned int* cg = cnt + bb;
    unsigned phase = 0;

    for (int t = 0; t < SEQ; ++t) {
        const int cur = t & 1, nxt = cur ^ 1;

        // ---- stage 1: h0' = tanh(xt@Wx0 + h0@Wh0 + b0) ----
        copy_async16(bA, h0b + (size_t)cur * BATCH * UNIT + rowBase * UNIT,
                     ROWS_WG * UNIT, tid);
        for (int idx = tid; idx < ROWS_WG * 32; idx += WG_THREADS) {
            int r = idx >> 5, c = idx & 31;
            int tok = tokens[(rowBase + r) * SEQ + t];
            float4 f4 = ((const float4*)(emb + (size_t)tok * EMB))[c];
            uint2 u;
            u.x = (unsigned)f2bf(f4.x) | ((unsigned)f2bf(f4.y) << 16);
            u.y = (unsigned)f2bf(f4.z) | ((unsigned)f2bf(f4.w) << 16);
            ((uint2*)s_xt)[r * 32 + c] = u;
        }
        wait_async0();
        __syncthreads();
        stage_compute<KT_E>(s_xt, EMB, wl_x0, s_A, wl_h0, aRow, hh,
                            h0b + (size_t)nxt * BATCH * UNIT, rowBase, mt, j, bias0);
        group_barrier(cg, ++phase);

        // ---- stage 2: h1' = tanh(h0'@Wx1 + h1@Wh1 + b1) ----
        copy_async16(bB, h0b + (size_t)nxt * BATCH * UNIT + rowBase * UNIT,
                     ROWS_WG * UNIT, tid);
        copy_async16(bA, h1b + (size_t)cur * BATCH * UNIT + rowBase * UNIT,
                     ROWS_WG * UNIT, tid);
        wait_async0();
        __syncthreads();
        stage_compute<KT_U>(s_B, UNIT, wl_x1, s_A, wl_h1, aRow, hh,
                            h1b + (size_t)nxt * BATCH * UNIT, rowBase, mt, j, bias1);
        group_barrier(cg, ++phase);

        // ---- stage 3: h2' = tanh(h1'@Wx1 + h2@Wh1 + b1) ----
        copy_async16(bA, h1b + (size_t)nxt * BATCH * UNIT + rowBase * UNIT,
                     ROWS_WG * UNIT, tid);
        copy_async16(bB, h2b + (size_t)cur * BATCH * UNIT + rowBase * UNIT,
                     ROWS_WG * UNIT, tid);
        wait_async0();
        __syncthreads();
        stage_compute<KT_U>(s_A, UNIT, wl_x1, s_B, wl_h1, aRow, hh,
                            h2b + (size_t)nxt * BATCH * UNIT, rowBase, mt, j, bias1);
        group_barrier(cg, ++phase);
    }

    // ---- FC head: out = sigmoid(h2 @ fc_w + fc_b), slice 0 only ----
    if (sl == 0) {
        const unsigned short* h2 = h2b;      // parity 0 after t=511
        int r = tid >> 3, sub = tid & 7;
        float acc = 0.f;
        for (int k = sub; k < UNIT; k += 8)
            acc += bf2f(h2[(rowBase + r) * UNIT + k]) * fcw[k];
        acc += __shfl_down(acc, 4, 8);
        acc += __shfl_down(acc, 2, 8);
        acc += __shfl_down(acc, 1, 8);
        if (sub == 0)
            out[rowBase + r] = 1.0f / (1.0f + __expf(-(acc + fcb[0])));
    }
}

extern "C" void kernel_launch(void* const* d_in, const int* in_sizes, int n_in,
                              void* d_out, int out_size, void* d_ws, size_t ws_size,
                              hipStream_t stream) {
    const int*   tokens = (const int*)  d_in[0];
    const float* emb    = (const float*)d_in[1];
    const float* Wx0    = (const float*)d_in[2];
    const float* Wh0    = (const float*)d_in[3];
    const float* b0     = (const float*)d_in[4];
    const float* Wx1    = (const float*)d_in[5];
    const float* Wh1    = (const float*)d_in[6];
    const float* b1     = (const float*)d_in[7];
    const float* fcw    = (const float*)d_in[8];
    const float* fcb    = (const float*)d_in[9];
    float* out = (float*)d_out;

    unsigned short* ws16 = (unsigned short*)d_ws;
    unsigned short* wfx0 = ws16;
    unsigned short* wfh0 = wfx0 + EMB * UNIT;
    unsigned short* wfx1 = wfh0 + UNIT * UNIT;
    unsigned short* wfh1 = wfx1 + UNIT * UNIT;
    unsigned short* hbuf = wfh1 + UNIT * UNIT;
    unsigned int*   cnt  = (unsigned int*)(hbuf + 6 * BATCH * UNIT);

    (void)in_sizes; (void)n_in; (void)out_size; (void)ws_size;

    hipFuncSetAttribute((const void*)rnn_main,
                        hipFuncAttributeMaxDynamicSharedMemorySize, SMEM_BYTES);

    const int zeroN = 6 * BATCH * UNIT / 2 + BBLOCKS;  // hbuf (as u32) + counters
    zero_ws<<<(zeroN + 255) / 256, 256, 0, stream>>>((unsigned int*)hbuf, zeroN);

    convert_weights<<<(EMB * UNIT) / 256, 256, 0, stream>>>(Wx0, wfx0, EMB);
    convert_weights<<<(UNIT * UNIT) / 256, 256, 0, stream>>>(Wh0, wfh0, UNIT);
    convert_weights<<<(UNIT * UNIT) / 256, 256, 0, stream>>>(Wx1, wfx1, UNIT);
    convert_weights<<<(UNIT * UNIT) / 256, 256, 0, stream>>>(Wh1, wfh1, UNIT);

    rnn_main<<<BBLOCKS * NSLICES, WG_THREADS, SMEM_BYTES, stream>>>(
        tokens, emb, b0, b1, fcw, fcb, wfx0, wfh0, wfx1, wfh1, hbuf, cnt, out);
}
